// VisibilityGNN_5858335392375
// MI455X (gfx1250) — matrix operand order repaired and checked by reading, not compiled, not run on hardware
//
#include <hip/hip_runtime.h>
#include <hip/hip_bf16.h>
#include <math.h>
#include <stdint.h>

// ---------------------------------------------------------------------------
// VisibilityGNN on MI455X (gfx1250, wave32).
//  - Layer GEMMs: v_wmma_f32_16x16x32_f16, B tiles resident in VGPRs,
//    4 row-tiles per block, fully unrolled K (templated 32/64).
//  - Edge gather/scatter: global_atomic_add_f32 + global_prefetch_b8.
//  - Heads: LDS-staged weights (one stage via global_load_async_to_lds_b32).
// ---------------------------------------------------------------------------

typedef __attribute__((ext_vector_type(16))) _Float16 v16h;
typedef __attribute__((ext_vector_type(8)))  float    v8f;

#define HID 64

// H[M,64] = act(X)[M,K] @ W[K,64] + b    (act = optional ReLU on input)
// block = 128 threads = 4 waves; wave w owns the 16-column tile [16w,16w+16).
// Each block processes TPB consecutive 16-row tiles with B resident in VGPRs.
template <int K, bool RELU_IN, int TPB>
__global__ void gemm_h_wmma(const float* __restrict__ X, const float* __restrict__ W,
                            const float* __restrict__ bias, float* __restrict__ H,
                            int M)
{
    constexpr int KT = K / 32;           // k-tiles of 32
    const int lane = threadIdx.x & 31;
    const int wave = threadIdx.x >> 5;
    const int n0   = wave * 16;
    const int m    = lane & 15;
    const int hi   = lane >> 4;          // half-wave id
    const int nc   = n0 + m;             // output column for B/C/D lanes

    // B tiles 32x16 f16, resident (lanes0-15: K=kt*32+[0..7|16..23], lanes16-31: +[8..15|24..31])
    v16h b[KT];
    #pragma unroll
    for (int kt = 0; kt < KT; ++kt) {
        const float* wc = W + (size_t)(kt * 32 + hi * 8) * HID + nc;
        #pragma unroll
        for (int e = 0; e < 16; ++e) b[kt][e] = (_Float16)wc[(e + (e & 8)) * HID];
    }

    const float bv = bias[nc];
    const int tile0 = blockIdx.x * TPB;

    #pragma unroll
    for (int t = 0; t < TPB; ++t) {
        const int row0 = (tile0 + t) * 16;
        if (row0 < M) {                  // block-uniform guard (EXEC stays full)
            // A tile 16x32 f16 (lanes0-15: K=[0..7|16..23], lanes16-31: K=[8..15|24..31])
            v8f acc = {};
            #pragma unroll
            for (int kt = 0; kt < KT; ++kt) {
                const float4* xr4 =
                    (const float4*)(X + (size_t)(row0 + m) * K + kt * 32 + hi * 8);
                float4 p0 = xr4[0], p1 = xr4[1];   // k = base .. base+7
                float4 q0 = xr4[4], q1 = xr4[5];   // k = base+16 .. base+23
                if (RELU_IN) {
                    p0.x = fmaxf(p0.x, 0.f); p0.y = fmaxf(p0.y, 0.f);
                    p0.z = fmaxf(p0.z, 0.f); p0.w = fmaxf(p0.w, 0.f);
                    p1.x = fmaxf(p1.x, 0.f); p1.y = fmaxf(p1.y, 0.f);
                    p1.z = fmaxf(p1.z, 0.f); p1.w = fmaxf(p1.w, 0.f);
                    q0.x = fmaxf(q0.x, 0.f); q0.y = fmaxf(q0.y, 0.f);
                    q0.z = fmaxf(q0.z, 0.f); q0.w = fmaxf(q0.w, 0.f);
                    q1.x = fmaxf(q1.x, 0.f); q1.y = fmaxf(q1.y, 0.f);
                    q1.z = fmaxf(q1.z, 0.f); q1.w = fmaxf(q1.w, 0.f);
                }
                v16h a;
                a[0]  = (_Float16)p0.x; a[1]  = (_Float16)p0.y;
                a[2]  = (_Float16)p0.z; a[3]  = (_Float16)p0.w;
                a[4]  = (_Float16)p1.x; a[5]  = (_Float16)p1.y;
                a[6]  = (_Float16)p1.z; a[7]  = (_Float16)p1.w;
                a[8]  = (_Float16)q0.x; a[9]  = (_Float16)q0.y;
                a[10] = (_Float16)q0.z; a[11] = (_Float16)q0.w;
                a[12] = (_Float16)q1.x; a[13] = (_Float16)q1.y;
                a[14] = (_Float16)q1.z; a[15] = (_Float16)q1.w;
                acc = __builtin_amdgcn_wmma_f32_16x16x32_f16(
                          false, a, false, b[kt], (short)0, acc, false, false);
            }
            #pragma unroll
            for (int v = 0; v < 8; ++v) {
                const int mr = v + hi * 8;     // C/D: VGPR v holds row v + 8*hi
                H[(size_t)(row0 + mr) * HID + nc] = acc[v] + bv;
            }
        }
    }
}

// ew[e] = sigmoid( relu(edge_attr[e] @ w1 + b1) @ w2 + b2 )
__global__ void edge_mlp_kernel(const float* __restrict__ ea,
                                const float* __restrict__ w1, const float* __restrict__ b1,
                                const float* __restrict__ w2, const float* __restrict__ b2,
                                float* __restrict__ ew, int E)
{
    int e = blockIdx.x * blockDim.x + threadIdx.x;
    if (e >= E) return;
    float a0 = ea[2 * e], a1 = ea[2 * e + 1];
    float z = b2[0];
    #pragma unroll
    for (int j = 0; j < 16; ++j) {
        float hj = fmaxf(a0 * w1[j] + a1 * w1[16 + j] + b1[j], 0.f);
        z += hj * w2[j];
    }
    ew[e] = 1.f / (1.f + expf(-z));
}

__global__ void zero_kernel(float* __restrict__ p, long long n)
{
    long long i = (long long)blockIdx.x * blockDim.x + threadIdx.x;
    if (i < n) p[i] = 0.f;
}

// out[dst[e]] += ew[e] * h[src[e]]   (16 threads per edge, float4 per thread)
__global__ void scatter_kernel(const float* __restrict__ h, const float* __restrict__ ew,
                               const int* __restrict__ src, const int* __restrict__ dst,
                               float* __restrict__ out, int E)
{
    long long t = (long long)blockIdx.x * blockDim.x + threadIdx.x;
    int e    = (int)(t >> 4);
    int part = (int)(t & 15);
    if (e >= E) return;
    // speculative prefetch of an upcoming gather row (global_prefetch_b8)
    if (part == 0) {
        int pe = e + 4096;
        if (pe < E) __builtin_prefetch(h + (size_t)src[pe] * HID, 0, 1);
    }
    int s = src[e], d = dst[e];
    float w = ew[e];
    float4 v = *((const float4*)(h + (size_t)s * HID) + part);
    float* op = out + (size_t)d * HID + part * 4;
    atomicAdd(op + 0, w * v.x);
    atomicAdd(op + 1, w * v.y);
    atomicAdd(op + 2, w * v.z);
    atomicAdd(op + 3, w * v.w);
}

// Fused regression + classification heads; weights staged in LDS.
__global__ void heads_kernel(const float* __restrict__ x,
    const float* __restrict__ rh1_w, const float* __restrict__ rh1_b,
    const float* __restrict__ rh2_w, const float* __restrict__ rh2_b,
    const float* __restrict__ mean_w, const float* __restrict__ mean_b,
    const float* __restrict__ std_w,  const float* __restrict__ std_b,
    const float* __restrict__ cls1_w, const float* __restrict__ cls1_b,
    const float* __restrict__ cls2_w, const float* __restrict__ cls2_b,
    float* __restrict__ out, int N)
{
    __shared__ float s_rh1[64 * 32];
    __shared__ float s_rh1b[32];
    __shared__ float s_rh2[32 * 16];
    __shared__ float s_rh2b[16];
    __shared__ float s_mw[16];
    __shared__ float s_sw[16];
    __shared__ float s_c1[64 * 32];
    __shared__ float s_c1b[32];
    __shared__ float s_c2[32 * 2];
    __shared__ float s_c2b[2];
    __shared__ float s_mb, s_sb;

    // Stage rh2_w via async global->LDS DMA (ASYNCcnt path, gfx1250).
    // VDST = LDS byte offset (low 32 bits of generic LDS pointer),
    // VADDR = per-lane byte offset, SADDR = 64-bit base.
    {
        unsigned lds0 = (unsigned)(uintptr_t)(&s_rh2[threadIdx.x]);
        unsigned lds1 = (unsigned)(uintptr_t)(&s_rh2[threadIdx.x + 256]);
        unsigned go0  = threadIdx.x * 4u;
        unsigned go1  = (threadIdx.x + 256u) * 4u;
        asm volatile("global_load_async_to_lds_b32 %0, %1, %2"
                     :: "v"(lds0), "v"(go0), "s"(rh2_w) : "memory");
        asm volatile("global_load_async_to_lds_b32 %0, %1, %2"
                     :: "v"(lds1), "v"(go1), "s"(rh2_w) : "memory");
    }

    for (int i = threadIdx.x; i < 64 * 32; i += blockDim.x) {
        s_rh1[i] = rh1_w[i];
        s_c1[i]  = cls1_w[i];
    }
    if (threadIdx.x < 32) { s_rh1b[threadIdx.x] = rh1_b[threadIdx.x];
                            s_c1b[threadIdx.x]  = cls1_b[threadIdx.x]; }
    if (threadIdx.x < 16) { s_rh2b[threadIdx.x] = rh2_b[threadIdx.x];
                            s_mw[threadIdx.x]   = mean_w[threadIdx.x];
                            s_sw[threadIdx.x]   = std_w[threadIdx.x]; }
    if (threadIdx.x < 64)   s_c2[threadIdx.x]   = cls2_w[threadIdx.x];
    if (threadIdx.x == 0) { s_c2b[0] = cls2_b[0]; s_c2b[1] = cls2_b[1];
                            s_mb = mean_b[0]; s_sb = std_b[0]; }

    asm volatile("s_wait_asynccnt 0" ::: "memory");
    __syncthreads();

    int n = blockIdx.x * blockDim.x + threadIdx.x;
    if (n >= N) return;

    float xi[64];
    #pragma unroll
    for (int c = 0; c < 64; ++c) xi[c] = fmaxf(x[(size_t)n * 64 + c], 0.f);

    float r1[32];
    for (int j = 0; j < 32; ++j) {
        float acc = s_rh1b[j];
        #pragma unroll
        for (int c = 0; c < 64; ++c) acc += xi[c] * s_rh1[c * 32 + j];
        r1[j] = fmaxf(acc, 0.f);
    }
    float r2[16];
    for (int j = 0; j < 16; ++j) {
        float acc = s_rh2b[j];
        #pragma unroll
        for (int c = 0; c < 32; ++c) acc += r1[c] * s_rh2[c * 16 + j];
        r2[j] = fmaxf(acc, 0.f);
    }
    float zm = s_mb, zs = s_sb;
    #pragma unroll
    for (int j = 0; j < 16; ++j) { zm += r2[j] * s_mw[j]; zs += r2[j] * s_sw[j]; }
    float sp = (zs > 20.f) ? zs : log1pf(expf(zs));   // softplus

    float l0 = s_c2b[0], l1 = s_c2b[1];
    for (int j = 0; j < 32; ++j) {
        float acc = s_c1b[j];
        #pragma unroll
        for (int c = 0; c < 64; ++c) acc += xi[c] * s_c1[c * 32 + j];
        float cj = fmaxf(acc, 0.f);
        l0 += cj * s_c2[j * 2 + 0];
        l1 += cj * s_c2[j * 2 + 1];
    }
    out[n]                 = zm;
    out[N + n]             = sp;
    out[2 * N + 2 * n]     = l0;
    out[2 * N + 2 * n + 1] = l1;
}

extern "C" void kernel_launch(void* const* d_in, const int* in_sizes, int n_in,
                              void* d_out, int out_size, void* d_ws, size_t ws_size,
                              hipStream_t stream)
{
    const float* x    = (const float*)d_in[0];
    const int*   ei   = (const int*)  d_in[1];
    const float* ea   = (const float*)d_in[2];
    const float* W0   = (const float*)d_in[3];
    const float* b0   = (const float*)d_in[4];
    const float* Ws   = (const float*)d_in[5];
    const float* bs   = (const float*)d_in[6];
    const float* ew1  = (const float*)d_in[7];
    const float* eb1  = (const float*)d_in[8];
    const float* ew2  = (const float*)d_in[9];
    const float* eb2  = (const float*)d_in[10];
    const float* rh1w = (const float*)d_in[11];
    const float* rh1b = (const float*)d_in[12];
    const float* rh2w = (const float*)d_in[13];
    const float* rh2b = (const float*)d_in[14];
    const float* mw   = (const float*)d_in[15];
    const float* mb   = (const float*)d_in[16];
    const float* sw   = (const float*)d_in[17];
    const float* sb   = (const float*)d_in[18];
    const float* c1w  = (const float*)d_in[19];
    const float* c1b  = (const float*)d_in[20];
    const float* c2w  = (const float*)d_in[21];
    const float* c2b  = (const float*)d_in[22];

    const int N = in_sizes[0] / 32;   // 100000 (divisible by 16)
    const int E = in_sizes[1] / 2;    // 1600000
    const int* srcI = ei;             // edge_index[0]
    const int* dstI = ei + E;         // edge_index[1]

    // workspace: h[N,64] | x_agg[N,64] | ew[E]   (~57.6 MB total)
    float* buf_h = (float*)d_ws;
    float* buf_x = buf_h + (size_t)N * HID;
    float* ewb   = buf_x + (size_t)N * HID;

    const long long xelems = (long long)N * HID;
    constexpr int TPB = 4;
    const int mtiles = N / 16;
    const int gblocks = (mtiles + TPB - 1) / TPB;

    for (int l = 0; l < 4; ++l) {
        if (l == 0) {
            gemm_h_wmma<32, false, TPB><<<gblocks, 128, 0, stream>>>(
                x, W0, b0, buf_h, N);
        } else {
            gemm_h_wmma<64, true, TPB><<<gblocks, 128, 0, stream>>>(
                buf_x, Ws + (size_t)(l - 1) * HID * HID,
                bs + (size_t)(l - 1) * HID, buf_h, N);
        }

        edge_mlp_kernel<<<(E + 255) / 256, 256, 0, stream>>>(
            ea, ew1 + (size_t)l * 32, eb1 + (size_t)l * 16,
            ew2 + (size_t)l * 16, eb2 + l, ewb, E);

        zero_kernel<<<(unsigned)((xelems + 255) / 256), 256, 0, stream>>>(buf_x, xelems);

        scatter_kernel<<<(unsigned)(((long long)E * 16 + 255) / 256), 256, 0, stream>>>(
            buf_h, ewb, srcI, dstI, buf_x, E);
    }

    heads_kernel<<<(N + 255) / 256, 256, 0, stream>>>(
        buf_x, rh1w, rh1b, rh2w, rh2b, mw, mb, sw, sb,
        c1w, c1b, c2w, c2b, (float*)d_out, N);
}
